// attentionWeightRecycle_63075889709525
// MI455X (gfx1250) — matrix operand order repaired and checked by reading, not compiled, not run on hardware
//
#include <hip/hip_runtime.h>
#include <hip/hip_bf16.h>
#include <math.h>

typedef __attribute__((ext_vector_type(16))) __bf16 v16bf;
typedef __attribute__((ext_vector_type(8)))  __bf16 v8bf;
typedef __attribute__((ext_vector_type(8)))  float  v8f;

union AFrag { v16bf v; v8bf h[2]; };

__device__ __forceinline__ unsigned short f32_to_bf16(float f) {
    unsigned int u = __float_as_uint(f);
    unsigned int r = u + 0x7fffu + ((u >> 16) & 1u);   // round-to-nearest-even
    return (unsigned short)(r >> 16);
}

__device__ __forceinline__ float bf16_to_f32(unsigned short u) {
    return __uint_as_float(((unsigned int)u) << 16);
}

__device__ __forceinline__ v8f zero_v8f() {
    v8f z = {0.f, 0.f, 0.f, 0.f, 0.f, 0.f, 0.f, 0.f};
    return z;
}

// ---------------- conversion kernels ----------------

__global__ void convert_bf16_kernel(const float* __restrict__ src,
                                    unsigned short* __restrict__ dst, int n) {
    int i = blockIdx.x * blockDim.x + threadIdx.x;
    if (i < n) dst[i] = f32_to_bf16(src[i]);
}

// weight [8][400][400] f32 -> [8][416][416] bf16, zero padded
__global__ void convert_weight_kernel(const float* __restrict__ w,
                                      unsigned short* __restrict__ dst) {
    int idx = blockIdx.x * blockDim.x + threadIdx.x;
    if (idx >= 8 * 416 * 416) return;
    int n   = idx / (416 * 416);
    int rem = idx - n * (416 * 416);
    int i   = rem / 416;
    int k   = rem - i * 416;
    float v = (i < 400 && k < 400) ? w[((size_t)n * 400 + i) * 400 + k] : 0.f;
    dst[idx] = f32_to_bf16(v);
}

// feature [8][256][80][80] -> fs bf16 [8][c=256][k=416] (4x4 block sums, K padded)
__global__ void pool_kernel(const float* __restrict__ feat,
                            unsigned short* __restrict__ fs) {
    int idx = blockIdx.x * blockDim.x + threadIdx.x;
    if (idx >= 8 * 256 * 416) return;
    int n   = idx / (256 * 416);
    int rem = idx - n * (256 * 416);
    int c   = rem / 416;
    int k   = rem - c * 416;
    float s = 0.f;
    if (k < 400) {
        int y0 = (k / 20) * 4, x0 = (k % 20) * 4;
        const float* base = feat + (((size_t)n * 256 + c) * 80 + y0) * 80 + x0;
#pragma unroll
        for (int dy = 0; dy < 4; ++dy)
#pragma unroll
            for (int dx = 0; dx < 4; ++dx)
                s += base[dy * 80 + dx];
    }
    fs[idx] = f32_to_bf16(s);
}

// ---------------- coarse attention GEMM: r[n,i,c] = sum_k weight[n,i,k]*fs[n,c,k] ----------------
__global__ __launch_bounds__(256) void gemm_coarse(
    const unsigned short* __restrict__ wbf, const unsigned short* __restrict__ fsbf,
    float* __restrict__ rbuf) {
    const int lane = threadIdx.x & 31;
    const int wave = threadIdx.x >> 5;
    const int n  = blockIdx.z;
    const int m0 = blockIdx.y * 32 + (wave >> 2) * 16;
    const int n0 = blockIdx.x * 128 + (wave & 3) * 32;
    const unsigned short* A = wbf  + (size_t)n * 416 * 416;
    const unsigned short* B = fsbf + (size_t)n * 256 * 416;
    const int arow = m0 + (lane & 15);
    const int koff = (lane >> 4) * 16;
    const int bl   = lane & 15;
    const int bk   = (lane >> 4) * 16;
    v8f acc0 = zero_v8f(), acc1 = zero_v8f();
    for (int k0 = 0; k0 < 416; k0 += 32) {
        AFrag a;
        a.h[0] = *(const v8bf*)(A + (size_t)arow * 416 + k0 + koff);
        a.h[1] = *(const v8bf*)(A + (size_t)arow * 416 + k0 + koff + 8);
        v16bf b0 = *(const v16bf*)(B + (size_t)(n0 + bl) * 416 + k0 + bk);
        v16bf b1 = *(const v16bf*)(B + (size_t)(n0 + 16 + bl) * 416 + k0 + bk);
        acc0 = __builtin_amdgcn_wmma_f32_16x16x32_bf16(false, a.v, false, b0, (short)0, acc0, false, false);
        acc1 = __builtin_amdgcn_wmma_f32_16x16x32_bf16(false, a.v, false, b1, (short)0, acc1, false, false);
    }
    const int rhalf = (lane >> 4) * 8;
#pragma unroll
    for (int v = 0; v < 8; ++v) {
        int i = m0 + v + rhalf;
        if (i < 400) {
            rbuf[((size_t)n * 400 + i) * 256 + n0 + bl]      = acc0[v];
            rbuf[((size_t)n * 400 + i) * 256 + n0 + 16 + bl] = acc1[v];
        }
    }
}

// ---------------- fused: residual + LN1 + MLP(GELU) + residual + LN2 ----------------
// 64-token tile, 256 threads (8 waves). Each wave: 32 rows (2 A-subtiles) x 64 cols
// (4 B-subtiles) per chunk -> every B fragment feeds 2 WMMAs.
// LDS (160 KB total -> 2 blocks/WGP):
//   region R [128 KB]: stage f32 (64KB) -> h bf16 (128KB) -> result f32 (64KB)
//   xb [32 KB]: bf16 post-LN1 activations (GEMM1 A operand + residual for LN2)
//   mu/rsd: tail 512B of R (only live while h is not)
#define TT 64

__global__ __launch_bounds__(256) void mlp_fused(
    const float* __restrict__ feat, const float* __restrict__ rbuf,
    const float* __restrict__ g1, const float* __restrict__ bln1,
    const unsigned short* __restrict__ w1b, const float* __restrict__ bias1,
    const unsigned short* __restrict__ w2b, const float* __restrict__ bias2,
    const float* __restrict__ g2, const float* __restrict__ bln2,
    float* __restrict__ out) {
    __shared__ __align__(32) char smem[163840];
    float*          xf  = (float*)smem;                      // 64 KB (stage / result)
    unsigned short* hb  = (unsigned short*)smem;             // 128 KB (hidden, bf16)
    unsigned short* xb  = (unsigned short*)(smem + 131072);  // 32 KB
    float*          mu  = (float*)(smem + 131072 - 512);     // 64 floats
    float*          rsd = mu + TT;                           // 64 floats

    const int tid  = threadIdx.x;
    const int lane = tid & 31;
    const int wave = tid >> 5;
    const int nimg = blockIdx.x / 100;
    const int t0   = (blockIdx.x % 100) * TT;

    // Stage: x = bhwc(feature) + broadcast(r)
    for (int idx = tid; idx < TT * 256; idx += 256) {
        int tt  = idx & (TT - 1);
        int c   = idx >> 6;
        int t   = t0 + tt;
        int row = t / 80;
        int col = t - row * 80;
        int ci  = (row >> 2) * 20 + (col >> 2);
        float v = feat[((size_t)nimg * 256 + c) * 6400 + t]
                + rbuf[((size_t)nimg * 400 + ci) * 256 + c];
        xf[tt * 256 + c] = v;
    }
    __syncthreads();
    if (tid < TT) {
        float s = 0.f, ss = 0.f;
        for (int c = 0; c < 256; ++c) { float v = xf[tid * 256 + c]; s += v; ss += v * v; }
        float m   = s * (1.0f / 256.0f);
        float var = ss * (1.0f / 256.0f) - m * m;
        mu[tid]  = m;
        rsd[tid] = rsqrtf(var + 1e-5f);
    }
    __syncthreads();
    for (int idx = tid; idx < TT * 256; idx += 256) {
        int tt = idx & (TT - 1);
        int c  = idx >> 6;
        float y = (xf[tt * 256 + c] - mu[tt]) * rsd[tt] * g1[c] + bln1[c];
        xb[tt * 256 + c] = f32_to_bf16(y);   // GEMM1 A operand + residual for LN2
    }
    __syncthreads();   // after this, region R is repurposed as hb

    const int mh   = wave >> 2;          // 0..1 -> rows mh*32 .. mh*32+31
    const int nq   = wave & 3;
    const int arow = mh * 32 + (lane & 15);
    const int koff = (lane >> 4) * 16;
    const int bl   = lane & 15;
    const int bk   = (lane >> 4) * 16;
    const int rv   = (lane >> 4) * 8;

    // GEMM1: [64x256] @ W1^T [256x1024], GELU, -> hb (bf16)
    for (int chunk = 0; chunk < 4; ++chunk) {
        const int n0 = chunk * 256 + nq * 64;
        v8f acc[2][4];
#pragma unroll
        for (int mi = 0; mi < 2; ++mi)
#pragma unroll
            for (int j = 0; j < 4; ++j) acc[mi][j] = zero_v8f();
        for (int k0 = 0; k0 < 256; k0 += 32) {
            AFrag a0, a1;
            a0.h[0] = *(const v8bf*)&xb[arow * 256 + k0 + koff];
            a0.h[1] = *(const v8bf*)&xb[arow * 256 + k0 + koff + 8];
            a1.h[0] = *(const v8bf*)&xb[(arow + 16) * 256 + k0 + koff];
            a1.h[1] = *(const v8bf*)&xb[(arow + 16) * 256 + k0 + koff + 8];
#pragma unroll
            for (int j = 0; j < 4; ++j) {
                const int cn = n0 + j * 16 + bl;
                v16bf b = *(const v16bf*)(w1b + (size_t)cn * 256 + k0 + bk);
                acc[0][j] = __builtin_amdgcn_wmma_f32_16x16x32_bf16(false, a0.v, false, b, (short)0, acc[0][j], false, false);
                acc[1][j] = __builtin_amdgcn_wmma_f32_16x16x32_bf16(false, a1.v, false, b, (short)0, acc[1][j], false, false);
            }
        }
#pragma unroll
        for (int mi = 0; mi < 2; ++mi)
#pragma unroll
            for (int j = 0; j < 4; ++j) {
                const int cn   = n0 + j * 16 + bl;
                const float bv = bias1[cn];
#pragma unroll
                for (int v = 0; v < 8; ++v) {
                    int rm  = mh * 32 + mi * 16 + v + rv;
                    float x = acc[mi][j][v] + bv;
                    x = 0.5f * x * (1.0f + erff(x * 0.70710678118654752f)); // exact GELU
                    hb[rm * 1024 + cn] = f32_to_bf16(x);
                }
            }
    }
    __syncthreads();

    // GEMM2: [64x1024] @ W2^T [1024x256]
    {
        const int n0 = nq * 64;
        v8f acc[2][4];
#pragma unroll
        for (int mi = 0; mi < 2; ++mi)
#pragma unroll
            for (int j = 0; j < 4; ++j) acc[mi][j] = zero_v8f();
        for (int k0 = 0; k0 < 1024; k0 += 32) {
            AFrag a0, a1;
            a0.h[0] = *(const v8bf*)&hb[arow * 1024 + k0 + koff];
            a0.h[1] = *(const v8bf*)&hb[arow * 1024 + k0 + koff + 8];
            a1.h[0] = *(const v8bf*)&hb[(arow + 16) * 1024 + k0 + koff];
            a1.h[1] = *(const v8bf*)&hb[(arow + 16) * 1024 + k0 + koff + 8];
#pragma unroll
            for (int j = 0; j < 4; ++j) {
                const int cn = n0 + j * 16 + bl;
                v16bf b = *(const v16bf*)(w2b + (size_t)cn * 1024 + k0 + bk);
                acc[0][j] = __builtin_amdgcn_wmma_f32_16x16x32_bf16(false, a0.v, false, b, (short)0, acc[0][j], false, false);
                acc[1][j] = __builtin_amdgcn_wmma_f32_16x16x32_bf16(false, a1.v, false, b, (short)0, acc[1][j], false, false);
            }
        }
        __syncthreads();   // everyone done reading hb; repurpose R as result f32
#pragma unroll
        for (int mi = 0; mi < 2; ++mi)
#pragma unroll
            for (int j = 0; j < 4; ++j) {
                const int cn   = n0 + j * 16 + bl;
                const float bv = bias2[cn];
#pragma unroll
                for (int v = 0; v < 8; ++v) {
                    int rm  = mh * 32 + mi * 16 + v + rv;
                    float y = acc[mi][j][v] + bv + bf16_to_f32(xb[rm * 256 + cn]);
                    xf[rm * 256 + cn] = y;
                }
            }
    }
    __syncthreads();
    if (tid < TT) {
        float s = 0.f, ss = 0.f;
        for (int c = 0; c < 256; ++c) { float v = xf[tid * 256 + c]; s += v; ss += v * v; }
        float m   = s * (1.0f / 256.0f);
        float var = ss * (1.0f / 256.0f) - m * m;
        mu[tid]  = m;
        rsd[tid] = rsqrtf(var + 1e-5f);
    }
    __syncthreads();
    for (int idx = tid; idx < TT * 256; idx += 256) {
        int c  = idx & 255;
        int tt = idx >> 8;
        float y = (xf[tt * 256 + c] - mu[tt]) * rsd[tt] * g2[c] + bln2[c];
        out[((size_t)nimg * 6400 + (t0 + tt)) * 256 + c] = y;
    }
}

extern "C" void kernel_launch(void* const* d_in, const int* in_sizes, int n_in,
                              void* d_out, int out_size, void* d_ws, size_t ws_size,
                              hipStream_t stream) {
    const float* weight  = (const float*)d_in[0];
    const float* feature = (const float*)d_in[1];
    const float* ln1_g   = (const float*)d_in[2];
    const float* ln1_b   = (const float*)d_in[3];
    const float* W1      = (const float*)d_in[4];
    const float* b1      = (const float*)d_in[5];
    const float* W2      = (const float*)d_in[6];
    const float* b2      = (const float*)d_in[7];
    const float* ln2_g   = (const float*)d_in[8];
    const float* ln2_b   = (const float*)d_in[9];
    float* out = (float*)d_out;

    char* ws  = (char*)d_ws;
    size_t off = 0;
    auto alloc = [&](size_t bytes) {
        void* p = ws + off;
        off = (off + bytes + 255) & ~(size_t)255;
        return p;
    };
    unsigned short* wbf  = (unsigned short*)alloc((size_t)8 * 416 * 416 * 2); // padded weight
    unsigned short* fsbf = (unsigned short*)alloc((size_t)8 * 256 * 416 * 2); // pooled feature (c-major)
    unsigned short* w1b  = (unsigned short*)alloc((size_t)1024 * 256 * 2);
    unsigned short* w2b  = (unsigned short*)alloc((size_t)256 * 1024 * 2);
    float*          rbuf = (float*)alloc((size_t)8 * 400 * 256 * 4);

    convert_weight_kernel<<<(8 * 416 * 416 + 255) / 256, 256, 0, stream>>>(weight, wbf);
    convert_bf16_kernel<<<(1024 * 256 + 255) / 256, 256, 0, stream>>>(W1, w1b, 1024 * 256);
    convert_bf16_kernel<<<(256 * 1024 + 255) / 256, 256, 0, stream>>>(W2, w2b, 256 * 1024);
    pool_kernel<<<(8 * 256 * 416 + 255) / 256, 256, 0, stream>>>(feature, fsbf);
    gemm_coarse<<<dim3(2, 13, 8), 256, 0, stream>>>(wbf, fsbf, rbuf);
    mlp_fused<<<800, 256, 0, stream>>>(feature, rbuf, ln1_g, ln1_b, w1b, b1,
                                       w2b, b2, ln2_g, ln2_b, out);
}
